// GraphTransformer_90237262889130
// MI455X (gfx1250) — matrix operand order repaired and checked by reading, not compiled, not run on hardware
//
#include <hip/hip_runtime.h>
#include <math.h>

typedef __bf16 bf16;
typedef __attribute__((ext_vector_type(16))) __bf16 v16bf;
typedef __attribute__((ext_vector_type(8)))  float  v8f;

#define DEV __device__ __forceinline__

constexpr int Bx = 2, Lx = 48, Nx = 3000, Dx = 128, LOUT = 24;
constexpr int BL    = Bx * Lx;       // 96
constexpr int NBLK  = BL + 1;        // 97 column blocks (96 bl-slices + pe)
constexpr int PITCH = NBLK * Dx;     // 12416
constexpr int FFD   = 4 * Dx;        // 512

// ---------------- WMMA helpers (CDNA5 16x16x32 bf16, wave32) ----------------
DEV v8f wmma_bf(v16bf a, v16bf b, v8f c) {
  return __builtin_amdgcn_wmma_f32_16x16x32_bf16(false, a, false, b, (short)0, c,
                                                 false, false);
}

// A fragment: A row-major (MxK); tile (m0,k0). Lane: row=lane&15, half=lane>>4.
// e in [0,16): K=k0+half*16+e
DEV v16bf ldA(const bf16* A, int lda, int m0, int k0) {
  int lane = threadIdx.x & 31, half = lane >> 4, r = lane & 15;
  const bf16* row = A + (m0 + r) * lda + k0 + half * 16;
  v16bf a;
#pragma unroll
  for (int e = 0; e < 8; ++e) a[e] = row[e];
#pragma unroll
  for (int e = 8; e < 16; ++e) a[e] = row[e];
  return a;
}
// B fragment, B row-major (KxN): lane col=lane&15; e: K=k0+half*16+e
DEV v16bf ldB(const bf16* Bm, int ldb, int k0, int n0) {
  int lane = threadIdx.x & 31, half = lane >> 4, c = lane & 15;
  const bf16* p = Bm + (k0 + half * 16) * ldb + n0 + c;
  v16bf b;
#pragma unroll
  for (int e = 0; e < 16; ++e) b[e] = p[e * ldb];
  return b;
}
// B fragment for X @ W^T with W row-major (NxK): contiguous 16 bf16 per lane
DEV v16bf ldBT(const bf16* W, int ldw, int k0, int n0) {
  int lane = threadIdx.x & 31, half = lane >> 4, c = lane & 15;
  const bf16* p = W + (n0 + c) * ldw + k0 + half * 16;
  v16bf b;
#pragma unroll
  for (int e = 0; e < 16; ++e) b[e] = p[e];
  return b;
}
DEV v8f ldC_bf(const bf16* C, int ldc, int m0, int n0) {
  int lane = threadIdx.x & 31, half = lane >> 4, c = lane & 15;
  v8f f;
#pragma unroll
  for (int r = 0; r < 8; ++r) f[r] = (float)C[(m0 + r + 8 * half) * ldc + n0 + c];
  return f;
}
DEV void stC_f32(float* C, int ldc, int m0, int n0, v8f a) {
  int lane = threadIdx.x & 31, half = lane >> 4, c = lane & 15;
#pragma unroll
  for (int r = 0; r < 8; ++r) C[(m0 + r + 8 * half) * ldc + n0 + c] = a[r];
}
DEV void stC_bf(bf16* C, int ldc, int m0, int n0, v8f a) {
  int lane = threadIdx.x & 31, half = lane >> 4, c = lane & 15;
#pragma unroll
  for (int r = 0; r < 8; ++r) C[(m0 + r + 8 * half) * ldc + n0 + c] = (bf16)a[r];
}

// ---------------- small prep kernels ----------------
__global__ void k_cvt(const float* src, bf16* dst, int n) {
  int t = blockIdx.x * 256 + threadIdx.x;
  if (t < n) dst[t] = (bf16)src[t];
}

__global__ void k_mean(const float* x_enc, float* meanG) {
  int t = blockIdx.x * 256 + threadIdx.x;
  if (t >= Bx * Nx) return;
  int b = t / Nx, n = t % Nx;
  float s = 0.f;
  for (int l = 0; l < Lx; ++l) s += x_enc[(size_t)(b * Lx + l) * Nx + n];
  meanG[t] = s * (1.f / Lx);
}

// pe = pe_raw^T @ pos_w^T + pos_b, written as bf16 into column-block 96 of Xb
__global__ void k_pe(const float* pe_raw, const float* pos_w, const float* pos_b,
                     bf16* Xb) {
  int t = blockIdx.x * 256 + threadIdx.x;
  if (t >= Nx * Dx) return;
  int n = t >> 7, d = t & 127;
  float v = pos_b[d];
#pragma unroll
  for (int c = 0; c < 3; ++c) v += pe_raw[c * Nx + n] * pos_w[d * 3 + c];
  Xb[((size_t)BL * Nx + n) * Dx + d] = (bf16)v;
}

// ---------------- temporal transformer: 1 workgroup per (b,n) sequence ------
DEV void ln_rows(const bf16* src, bf16* dst, const float* g, const float* bb,
                 int tid) {
  if (tid < Lx) {
    const bf16* r = src + tid * Dx;
    float m = 0.f;
    for (int d = 0; d < Dx; ++d) m += (float)r[d];
    m *= (1.f / Dx);
    float v = 0.f;
    for (int d = 0; d < Dx; ++d) { float x = (float)r[d] - m; v += x * x; }
    float rs = rsqrtf(v * (1.f / Dx) + 1e-5f);
    bf16* o = dst + tid * Dx;
    for (int d = 0; d < Dx; ++d) o[d] = (bf16)(((float)r[d] - m) * rs * g[d] + bb[d]);
  }
}

__global__ __launch_bounds__(256)
void k_xformer(const float* x_enc, const float* meanG, const float* conv_w,
               const bf16* wq, const bf16* wk, const bf16* wv, const bf16* wo,
               const float* ln1_g, const float* ln1_b,
               const bf16* fw1, const float* fb1, const bf16* fw2, const float* fb2,
               const float* ln2_g, const float* ln2_b, bf16* Xb) {
  // 60KB phase-overlaid LDS arena
  __shared__ __align__(16) char smem[61440];
  bf16*  encS = (bf16*)(smem);           // [0,12288)   48x128 residual (bf16)
  bf16*  hS   = (bf16*)(smem + 12288);   // [12288,24576) LN output
  float* xm   = (float*)(smem + 12288);  // phase a only (overlays hS)
  float* T    = (float*)(smem + 24576);  // [24576,49152) f32 48x128 scratch
  bf16*  kS   = (bf16*)(smem + 24576);   // 48x128 (after T dead)
  bf16*  vS   = (bf16*)(smem + 24576);   // 64x128 (after kS dead)
  bf16*  a2   = (bf16*)(smem + 40960);   // 48x64 attn scores (bf16)
  bf16*  qS   = (bf16*)(smem + 49152);   // 48x128
  bf16*  gS   = (bf16*)(smem + 49152);   // reuse of qS region

  int bn = blockIdx.x, b = bn / Nx, n = bn % Nx;
  int tid = threadIdx.x, wave = tid >> 5, lane = tid & 31;
  int half = lane >> 4, lc = lane & 15;
  int n0 = wave * 16;

  // a: token embedding (circular conv1d k=3)
  if (tid < Lx) xm[tid] = x_enc[(size_t)(b * Lx + tid) * Nx + n] - meanG[b * Nx + n];
  __syncthreads();
  for (int idx = tid; idx < Lx * Dx; idx += 256) {
    int l = idx >> 7, d = idx & 127;
    float v = conv_w[d * 3 + 0] * xm[(l + Lx - 1) % Lx] +
              conv_w[d * 3 + 1] * xm[l] +
              conv_w[d * 3 + 2] * xm[(l + 1) % Lx];
    encS[idx] = (bf16)v;
  }
  __syncthreads();

  // b: LN1
  ln_rows(encS, hS, ln1_g, ln1_b, tid);
  __syncthreads();

  // c: q = softmax(h wq^T, dim=-1) * d^-0.5
  for (int mi = 0; mi < 3; ++mi) {
    v8f acc = {};
#pragma unroll
    for (int k0 = 0; k0 < 128; k0 += 32)
      acc = wmma_bf(ldA(hS, Dx, mi * 16, k0), ldBT(wq, Dx, k0, n0), acc);
    stC_f32(T, Dx, mi * 16, n0, acc);
  }
  __syncthreads();
  if (tid < Lx) {
    float* r = T + tid * Dx;
    float mx = r[0];
    for (int d = 1; d < Dx; ++d) mx = fmaxf(mx, r[d]);
    float sum = 0.f;
    for (int d = 0; d < Dx; ++d) sum += expf(r[d] - mx);
    float inv = 0.08838834764831845f / sum;  // 128^-0.5 / sum
    bf16* o = qS + tid * Dx;
    for (int d = 0; d < Dx; ++d) o[d] = (bf16)(expf(r[d] - mx) * inv);
  }
  __syncthreads();

  // d: k = softmax over L, column-softmax done fully in C-fragments + shfl
  {
    v8f zero = {};
    v8f acc[3] = {zero, zero, zero};
    for (int mi = 0; mi < 3; ++mi)
#pragma unroll
      for (int k0 = 0; k0 < 128; k0 += 32)
        acc[mi] = wmma_bf(ldA(hS, Dx, mi * 16, k0), ldBT(wk, Dx, k0, n0), acc[mi]);
    float mx = -1e30f;
    for (int mi = 0; mi < 3; ++mi)
#pragma unroll
      for (int r = 0; r < 8; ++r) mx = fmaxf(mx, acc[mi][r]);
    mx = fmaxf(mx, __shfl_xor(mx, 16));
    float sum = 0.f;
    for (int mi = 0; mi < 3; ++mi)
#pragma unroll
      for (int r = 0; r < 8; ++r) { float e = expf(acc[mi][r] - mx); acc[mi][r] = e; sum += e; }
    sum += __shfl_xor(sum, 16);
    float inv = 1.f / sum;
    for (int mi = 0; mi < 3; ++mi)
#pragma unroll
      for (int r = 0; r < 8; ++r)
        kS[(mi * 16 + r + 8 * half) * Dx + n0 + lc] = (bf16)(acc[mi][r] * inv);
  }
  __syncthreads();

  // f: att2 = q @ k^T  (48x48, zero-padded to 48x64 bf16)
  for (int idx = tid; idx < 48 * 64; idx += 256) a2[idx] = (bf16)0.f;
  __syncthreads();
  if (wave < 3) {
    int an0 = wave * 16;
    for (int mi = 0; mi < 3; ++mi) {
      v8f acc = {};
#pragma unroll
      for (int k0 = 0; k0 < 128; k0 += 32)
        acc = wmma_bf(ldA(qS, Dx, mi * 16, k0), ldBT(kS, Dx, k0, an0), acc);
      stC_bf(a2, 64, mi * 16, an0, acc);
    }
  }
  __syncthreads();

  // e: v = h wv^T (bf16, pad rows 48..63 = 0)
  for (int idx = tid; idx < 16 * Dx; idx += 256) vS[48 * Dx + idx] = (bf16)0.f;
  for (int mi = 0; mi < 3; ++mi) {
    v8f acc = {};
#pragma unroll
    for (int k0 = 0; k0 < 128; k0 += 32)
      acc = wmma_bf(ldA(hS, Dx, mi * 16, k0), ldBT(wv, Dx, k0, n0), acc);
    stC_bf(vS, Dx, mi * 16, n0, acc);
  }
  __syncthreads();

  // g: core = att2 @ v  (48x128)
  for (int mi = 0; mi < 3; ++mi) {
    v8f acc = {};
#pragma unroll
    for (int k0 = 0; k0 < 64; k0 += 32)
      acc = wmma_bf(ldA(a2, 64, mi * 16, k0), ldB(vS, Dx, k0, n0), acc);
    stC_bf(gS, Dx, mi * 16, n0, acc);
  }
  __syncthreads();

  // i: enc += core @ wo^T (residual accumulated in the C operand)
  for (int mi = 0; mi < 3; ++mi) {
    v8f acc = ldC_bf(encS, Dx, mi * 16, n0);
#pragma unroll
    for (int k0 = 0; k0 < 128; k0 += 32)
      acc = wmma_bf(ldA(gS, Dx, mi * 16, k0), ldBT(wo, Dx, k0, n0), acc);
    stC_bf(encS, Dx, mi * 16, n0, acc);
  }
  __syncthreads();

  // j: LN2
  ln_rows(encS, hS, ln2_g, ln2_b, tid);
  __syncthreads();

  // k: FF 128->512(gelu)->128, persistent accumulators across 4 j-blocks
  v8f accF[3];
  for (int mi = 0; mi < 3; ++mi) accF[mi] = ldC_bf(encS, Dx, mi * 16, n0);
  for (int jb = 0; jb < 4; ++jb) {
    for (int mi = 0; mi < 3; ++mi) {
      v8f acc = {};
#pragma unroll
      for (int k0 = 0; k0 < 128; k0 += 32)
        acc = wmma_bf(ldA(hS, Dx, mi * 16, k0), ldBT(fw1, Dx, k0, jb * 128 + n0), acc);
      stC_f32(T, Dx, mi * 16, n0, acc);
    }
    __syncthreads();
    for (int idx = tid; idx < Lx * Dx; idx += 256) {
      int d = idx & 127;
      float x = T[idx] + fb1[jb * 128 + d];
      gS[idx] = (bf16)(0.5f * x * (1.f + erff(x * 0.70710678118654752f)));
    }
    __syncthreads();
    for (int mi = 0; mi < 3; ++mi)
#pragma unroll
      for (int k0 = 0; k0 < 128; k0 += 32)
        accF[mi] = wmma_bf(ldA(gS, Dx, mi * 16, k0),
                           ldBT(fw2, FFD, jb * 128 + k0, n0), accF[mi]);
    __syncthreads();
  }
  // store (enc + ff + fb2) as bf16 into Xb layout [bl][n][d]
  for (int mi = 0; mi < 3; ++mi)
#pragma unroll
    for (int r = 0; r < 8; ++r) {
      int l = mi * 16 + r + 8 * half;
      Xb[((size_t)(b * Lx + l) * Nx + n) * Dx + n0 + lc] =
          (bf16)(accF[mi][r] + fb2[n0 + lc]);
    }
}

// ---------------- adjacency GEMM: Yb[m, bl*128+d] = adj @ Xb ----------------
__global__ __launch_bounds__(256)
void k_adj(const bf16* adjb, const bf16* Xb, bf16* Yb) {
  __shared__ bf16 As[64 * 72];    // 64x64 adj tile, padded pitch
  __shared__ bf16 Bs[64 * 136];   // 64x128 X tile, padded pitch
  int m0 = blockIdx.x * 64;
  int bl = blockIdx.y;            // 0..96 (96 == pe block)
  int tid = threadIdx.x, wave = tid >> 5, lane = tid & 31;
  int mi = wave & 3, nb = (wave >> 2) * 64;
  v8f zero = {};
  v8f acc[4] = {zero, zero, zero, zero};
  for (int k0 = 0; k0 < 3008; k0 += 64) {
#pragma unroll
    for (int i = 0; i < 2; ++i) {  // A: 64x64 = 512 uint4
      int v = tid + i * 256, r = v >> 3, cv = (v & 7) * 8;
      uint4 z = {0u, 0u, 0u, 0u};
      if (m0 + r < Nx && k0 + cv < Nx)
        z = *(const uint4*)(adjb + (size_t)(m0 + r) * Nx + k0 + cv);
      *(uint4*)(As + r * 72 + cv) = z;
    }
#pragma unroll
    for (int i = 0; i < 4; ++i) {  // B: 64x128 = 1024 uint4
      int v = tid + i * 256, r = v >> 4, cv = (v & 15) * 8;
      uint4 z = {0u, 0u, 0u, 0u};
      if (k0 + r < Nx)
        z = *(const uint4*)(Xb + ((size_t)bl * Nx + k0 + r) * Dx + cv);
      *(uint4*)(Bs + r * 136 + cv) = z;
    }
    __syncthreads();
#pragma unroll
    for (int kk = 0; kk < 64; kk += 32) {
      v16bf a = ldA(As, 72, mi * 16, kk);
#pragma unroll
      for (int ni = 0; ni < 4; ++ni)
        acc[ni] = wmma_bf(a, ldB(Bs, 136, kk, nb + ni * 16), acc[ni]);
    }
    __syncthreads();
  }
  int half = lane >> 4, c = lane & 15;
#pragma unroll
  for (int ni = 0; ni < 4; ++ni)
#pragma unroll
    for (int r = 0; r < 8; ++r) {
      int m = m0 + mi * 16 + r + 8 * half;
      if (m < Nx)
        Yb[(size_t)m * PITCH + bl * Dx + nb + ni * 16 + c] = (bf16)acc[ni][r];
    }
}

// ---------------- pw = pe2 @ weights_pool (3000 x 16384) --------------------
__global__ __launch_bounds__(256)
void k_pw(const bf16* Yb, const bf16* Wp, bf16* pw) {
  __shared__ bf16 As[64 * 136];
  __shared__ bf16 Bs[128 * 136];
  int m0 = blockIdx.x * 64, nblk = blockIdx.y;  // nblk 0..127
  int tid = threadIdx.x, wave = tid >> 5, lane = tid & 31;
  int mi = wave & 3, nb = (wave >> 2) * 64;
#pragma unroll
  for (int i = 0; i < 4; ++i) {  // pe2 tile: 64x128
    int v = tid + i * 256, r = v >> 4, cv = (v & 15) * 8;
    uint4 z = {0u, 0u, 0u, 0u};
    if (m0 + r < Nx)
      z = *(const uint4*)(Yb + (size_t)(m0 + r) * PITCH + BL * Dx + cv);
    *(uint4*)(As + r * 136 + cv) = z;
  }
#pragma unroll
  for (int i = 0; i < 8; ++i) {  // Wp block: 128x128
    int v = tid + i * 256, r = v >> 4, cv = (v & 15) * 8;
    *(uint4*)(Bs + r * 136 + cv) =
        *(const uint4*)(Wp + (size_t)r * 16384 + nblk * 128 + cv);
  }
  __syncthreads();
  v8f zero = {};
  v8f acc[4] = {zero, zero, zero, zero};
#pragma unroll
  for (int k0 = 0; k0 < 128; k0 += 32) {
    v16bf a = ldA(As, 136, mi * 16, k0);
#pragma unroll
    for (int ni = 0; ni < 4; ++ni)
      acc[ni] = wmma_bf(a, ldB(Bs, 136, k0, nb + ni * 16), acc[ni]);
  }
  int half = lane >> 4, c = lane & 15;
#pragma unroll
  for (int ni = 0; ni < 4; ++ni)
#pragma unroll
    for (int r = 0; r < 8; ++r) {
      int m = m0 + mi * 16 + r + 8 * half;
      if (m < Nx)
        pw[(size_t)m * 16384 + nblk * 128 + nb + ni * 16 + c] = (bf16)acc[ni][r];
    }
}

// ---------------- per-node grouped GEMM: Z[bl,n,o] = Y_n @ pw_n -------------
__global__ __launch_bounds__(256)
void k_group(const bf16* Yb, const bf16* pw, float* Z) {
  __shared__ bf16 As[96 * 136];
  __shared__ bf16 Bs[128 * 136];
  int n = blockIdx.x;
  int tid = threadIdx.x, wave = tid >> 5, lane = tid & 31;
#pragma unroll
  for (int i = 0; i < 6; ++i) {  // A: 96x128 (bl x i) — contiguous in Yb row n
    int v = tid + i * 256, r = v >> 4, cv = (v & 15) * 8;
    *(uint4*)(As + r * 136 + cv) =
        *(const uint4*)(Yb + (size_t)n * PITCH + r * Dx + cv);
  }
#pragma unroll
  for (int i = 0; i < 8; ++i) {  // B: pw_n 128x128
    int v = tid + i * 256, r = v >> 4, cv = (v & 15) * 8;
    *(uint4*)(Bs + r * 136 + cv) =
        *(const uint4*)(pw + (size_t)n * 16384 + r * 128 + cv);
  }
  __syncthreads();
  v8f zero = {};
  v8f acc[6] = {zero, zero, zero, zero, zero, zero};
#pragma unroll
  for (int k0 = 0; k0 < 128; k0 += 32) {
    v16bf b = ldB(Bs, 136, k0, wave * 16);
#pragma unroll
    for (int mi = 0; mi < 6; ++mi)
      acc[mi] = wmma_bf(ldA(As, 136, mi * 16, k0), b, acc[mi]);
  }
  int half = lane >> 4, c = lane & 15;
#pragma unroll
  for (int mi = 0; mi < 6; ++mi)
#pragma unroll
    for (int r = 0; r < 8; ++r) {
      int bl = mi * 16 + r + 8 * half;
      Z[((size_t)bl * Nx + n) * Dx + wave * 16 + c] = acc[mi][r];
    }
}

// ---------------- GroupNorm stats per (bl, group) ---------------------------
__global__ __launch_bounds__(256)
void k_gnstats(const float* Z, float* stats) {
  __shared__ float red[4 * 256];
  int bl = blockIdx.x, tid = threadIdx.x;
  const float* base = Z + (size_t)bl * Nx * Dx;
  float s[4] = {0, 0, 0, 0}, q[4] = {0, 0, 0, 0};
  for (int idx = tid; idx < Nx * Dx; idx += 256) {
    float v = base[idx];
    int g = (idx >> 5) & 3;
    s[g] += v; q[g] += v * v;
  }
#pragma unroll
  for (int g = 0; g < 4; ++g) red[g * 256 + tid] = s[g];
  __syncthreads();
  for (int off = 128; off; off >>= 1) {
    if (tid < off)
#pragma unroll
      for (int g = 0; g < 4; ++g) red[g * 256 + tid] += red[g * 256 + tid + off];
    __syncthreads();
  }
  if (tid < 4) stats[bl * 8 + tid] = red[tid * 256] * (1.f / (Nx * 32));
  __syncthreads();
#pragma unroll
  for (int g = 0; g < 4; ++g) red[g * 256 + tid] = q[g];
  __syncthreads();
  for (int off = 128; off; off >>= 1) {
    if (tid < off)
#pragma unroll
      for (int g = 0; g < 4; ++g) red[g * 256 + tid] += red[g * 256 + tid + off];
    __syncthreads();
  }
  if (tid < 4) {
    float mu = stats[bl * 8 + tid];
    float var = red[tid * 256] * (1.f / (Nx * 32)) - mu * mu;
    stats[bl * 8 + 4 + tid] = rsqrtf(var + 1e-5f);
  }
}

// ---------------- normalize + p2 projection + SiLU --------------------------
__global__ __launch_bounds__(256)
void k_s(const float* Z, const float* stats, const float* gn_g, const float* gn_b,
         const float* p2_w, const float* p2_b, float* sOut) {
  int bl = blockIdx.y, wave = threadIdx.x >> 5, lane = threadIdx.x & 31;
  int n = blockIdx.x * 8 + wave;  // Nx = 375*8
  const float* z = Z + ((size_t)bl * Nx + n) * Dx;
  const float* st = stats + bl * 8;
  float p = 0.f;
#pragma unroll
  for (int j = 0; j < 4; ++j) {
    int d = lane * 4 + j, g = d >> 5;
    float v = (z[d] - st[g]) * st[4 + g] * gn_g[d] + gn_b[d];
    p += v * p2_w[d];
  }
#pragma unroll
  for (int off = 16; off; off >>= 1) p += __shfl_xor(p, off);
  if (lane == 0) {
    float t = p + p2_b[0];
    sOut[(size_t)bl * Nx + n] = t / (1.f + expf(-t));
  }
}

// ---------------- temporal head p1 + mean residual --------------------------
__global__ void k_final(const float* s, const float* p1_w, const float* p1_b,
                        const float* meanG, float* out) {
  int t = blockIdx.x * 256 + threadIdx.x;
  if (t >= Bx * LOUT * Nx) return;
  int n = t % Nx, lo = (t / Nx) % LOUT, b = t / (Nx * LOUT);
  float acc = p1_b[lo];
  for (int l = 0; l < Lx; ++l)
    acc += p1_w[lo * Lx + l] * s[(size_t)(b * Lx + l) * Nx + n];
  out[t] = acc + meanG[b * Nx + n];
}

// ---------------- host launcher ---------------------------------------------
extern "C" void kernel_launch(void* const* d_in, const int* in_sizes, int n_in,
                              void* d_out, int out_size, void* d_ws, size_t ws_size,
                              hipStream_t stream) {
  (void)in_sizes; (void)n_in; (void)out_size; (void)ws_size;
  const float* x_enc  = (const float*)d_in[0];
  const float* adj    = (const float*)d_in[1];
  const float* pe_raw = (const float*)d_in[2];
  const float* conv_w = (const float*)d_in[3];
  const float* pos_w  = (const float*)d_in[4];
  const float* pos_b  = (const float*)d_in[5];
  const float* wq     = (const float*)d_in[6];
  const float* wk     = (const float*)d_in[7];
  const float* wv     = (const float*)d_in[8];
  const float* wo     = (const float*)d_in[9];
  const float* ln1_g  = (const float*)d_in[10];
  const float* ln1_b  = (const float*)d_in[11];
  const float* ff_w1  = (const float*)d_in[12];
  const float* ff_b1  = (const float*)d_in[13];
  const float* ff_w2  = (const float*)d_in[14];
  const float* ff_b2  = (const float*)d_in[15];
  const float* ln2_g  = (const float*)d_in[16];
  const float* ln2_b  = (const float*)d_in[17];
  const float* wpool  = (const float*)d_in[18];
  const float* gn_g   = (const float*)d_in[19];
  const float* gn_b   = (const float*)d_in[20];
  const float* p2_w   = (const float*)d_in[21];
  const float* p2_b   = (const float*)d_in[22];
  const float* p1_w   = (const float*)d_in[23];
  const float* p1_b   = (const float*)d_in[24];
  float* out = (float*)d_out;

  char* ws = (char*)d_ws;
  size_t off = 0;
  auto alloc = [&](size_t bytes) {
    void* p = ws + off;
    off = (off + bytes + 255) & ~(size_t)255;
    return p;
  };
  bf16* adjb  = (bf16*)alloc((size_t)Nx * Nx * 2);
  bf16* Xb    = (bf16*)alloc((size_t)NBLK * Nx * Dx * 2);
  bf16* Yb    = (bf16*)alloc((size_t)Nx * PITCH * 2);
  bf16* wq_b  = (bf16*)alloc(Dx * Dx * 2);
  bf16* wk_b  = (bf16*)alloc(Dx * Dx * 2);
  bf16* wv_b  = (bf16*)alloc(Dx * Dx * 2);
  bf16* wo_b  = (bf16*)alloc(Dx * Dx * 2);
  bf16* fw1_b = (bf16*)alloc(FFD * Dx * 2);
  bf16* fw2_b = (bf16*)alloc(Dx * FFD * 2);
  bf16* Wp_b  = (bf16*)alloc((size_t)Dx * Dx * Dx * 2);
  bf16* pw_b  = (bf16*)alloc((size_t)Nx * Dx * Dx * 2);
  float* Z    = (float*)alloc((size_t)BL * Nx * Dx * 4);
  float* meanG = (float*)alloc((size_t)Bx * Nx * 4);
  float* stats = (float*)alloc((size_t)BL * 8 * 4);
  float* sBuf  = (float*)alloc((size_t)BL * Nx * 4);

  auto cdiv = [](int a, int b) { return (a + b - 1) / b; };

  k_cvt<<<cdiv(Nx * Nx, 256), 256, 0, stream>>>(adj, adjb, Nx * Nx);
  k_cvt<<<cdiv(Dx * Dx, 256), 256, 0, stream>>>(wq, wq_b, Dx * Dx);
  k_cvt<<<cdiv(Dx * Dx, 256), 256, 0, stream>>>(wk, wk_b, Dx * Dx);
  k_cvt<<<cdiv(Dx * Dx, 256), 256, 0, stream>>>(wv, wv_b, Dx * Dx);
  k_cvt<<<cdiv(Dx * Dx, 256), 256, 0, stream>>>(wo, wo_b, Dx * Dx);
  k_cvt<<<cdiv(FFD * Dx, 256), 256, 0, stream>>>(ff_w1, fw1_b, FFD * Dx);
  k_cvt<<<cdiv(Dx * FFD, 256), 256, 0, stream>>>(ff_w2, fw2_b, Dx * FFD);
  k_cvt<<<cdiv(Dx * Dx * Dx, 256), 256, 0, stream>>>(wpool, Wp_b, Dx * Dx * Dx);
  k_mean<<<cdiv(Bx * Nx, 256), 256, 0, stream>>>(x_enc, meanG);
  k_pe<<<cdiv(Nx * Dx, 256), 256, 0, stream>>>(pe_raw, pos_w, pos_b, Xb);

  k_xformer<<<Bx * Nx, 256, 0, stream>>>(x_enc, meanG, conv_w,
                                         wq_b, wk_b, wv_b, wo_b, ln1_g, ln1_b,
                                         fw1_b, ff_b1, fw2_b, ff_b2, ln2_g, ln2_b,
                                         Xb);
  k_adj<<<dim3(cdiv(Nx, 64), NBLK), 256, 0, stream>>>(adjb, Xb, Yb);
  k_pw<<<dim3(cdiv(Nx, 64), Dx), 256, 0, stream>>>(Yb, Wp_b, pw_b);
  k_group<<<Nx, 256, 0, stream>>>(Yb, pw_b, Z);
  k_gnstats<<<BL, 256, 0, stream>>>(Z, stats);
  k_s<<<dim3(Nx / 8, BL), 256, 0, stream>>>(Z, stats, gn_g, gn_b, p2_w, p2_b, sBuf);
  k_final<<<cdiv(Bx * LOUT * Nx, 256), 256, 0, stream>>>(sBuf, p1_w, p1_b, meanG, out);
}
